// DAHead_21792664060051
// MI455X (gfx1250) — matrix operand order repaired and checked by reading, not compiled, not run on hardware
//
#include <hip/hip_runtime.h>
#include <hip/hip_bf16.h>

// ---------------------------------------------------------------------------
// DANet dual-attention head for MI455X (gfx1250, wave32, WMMA).
// v6: im2col K reordered to (ky,kx,ci) + matching weight repack ->
//     no per-element division, chunk-constant border predicate, and
//     immediate-offset global loads (hw/w shifts are template params).
//     Plus all of v5: pk_mul masking, v8h LDS stores, guard-free GEMM,
//     register-staged prefetch overlapping HBM with the matrix pipe.
// ---------------------------------------------------------------------------

#include <stdint.h>

typedef _Float16 h16;
typedef __attribute__((ext_vector_type(16))) _Float16 v16h;
typedef __attribute__((ext_vector_type(8)))  _Float16 v8h;
typedef __attribute__((ext_vector_type(8)))  float    v8f;

#define BN_EPS 1e-5f

// A-matrix fragment (16x32 f16), contiguous K order (matches B):
// lanes 0-15: K[0..15] in halves 0-15
// lanes 16-31: K[16..31]
__device__ static inline v16h lds_load_a_frag(const h16* arow, int lane) {
    int base0 = (lane >> 4) << 4;              // 0 or 16 halves
    v8h lo = *(const v8h*)(arow + base0);
    v8h hi = *(const v8h*)(arow + 8 + base0);
    return __builtin_shufflevector(lo, hi, 0, 1, 2, 3, 4, 5, 6, 7,
                                   8, 9, 10, 11, 12, 13, 14, 15);
}

#define WMMA_F16(a, b, c) \
    __builtin_amdgcn_wmma_f32_16x16x32_f16(false, (a), false, (b), (short)0, \
                                           (c), false, false)

// ---------------------------------------------------------------------------
// Implicit-GEMM conv (K3: 3x3 pad 1, else 1x1 pad 0). HWS/WS = log2(H*W),
// log2(W) as template constants -> immediate-offset loads.
// K order is (ky,kx,ci): k = (ky*3+kx)*Cin + ci. Cin is a power of two and
// a multiple of 32, so each 32-K chunk has a single uniform (ky,kx) and a
// chunk-constant border predicate per thread.
// Weights packed to match: wp[co*K + (ky*3+kx)*Cin + ci], rows zero-padded
// to a multiple of 64 output channels.
// Block 256 thr = 8 waves (2x4); tile M=64, N=128, K-chunk 32; each wave
// computes a 32x32 output tile = 4 WMMA per chunk.
//   A: row mi = t&63, K-run ka = (t>>6)*8  -> one v8h global load + store
//   B: col nl = t&127, K-run kb = (t>>7)*16 -> two v8h stores
// MGUARD: epilogue channel guard (conv4b). BN+ReLU iff scale != nullptr.
// ---------------------------------------------------------------------------
template <bool MGUARD, bool K3, int HWS, int WS>
__global__ __launch_bounds__(256)
void conv_implicit_wmma(const h16* __restrict__ in, const h16* __restrict__ wp,
                        const float* __restrict__ scale,
                        const float* __restrict__ shift,
                        h16* __restrict__ out16, float* __restrict__ out32,
                        int Cin, int cin_shift, int Cout) {
    __shared__ h16 Als[64 * 32] __attribute__((aligned(32)));
    __shared__ h16 Bls[128 * 32] __attribute__((aligned(32)));

    constexpr int HW = 1 << HWS;
    constexpr int W = 1 << WS;
    constexpr int H = 1 << (HWS - WS);

    const int t = threadIdx.x;
    const int lane = t & 31;
    const int wave = t >> 5;
    const int wr = wave >> 2;   // 0..1 -> 32-row block
    const int wc = wave & 3;    // 0..3 -> 32-col block
    const int K = K3 ? Cin * 9 : Cin;
    const int n0 = blockIdx.x * 128;
    const int m0 = blockIdx.y * 64;

    // per-thread constant tile coordinates
    const int mi = t & 63;
    const int ka = (t >> 6) << 3;     // A K-run base (0,8,16,24)
    const int nl = t & 127;
    const int kb = (t >> 7) << 4;     // B K-run base (0,16)
    const uint32_t arow = (uint32_t)(m0 + mi) * (uint32_t)K;
    const int ng = n0 + nl;
    const int bb = ng >> HWS;
    const int p = ng & (HW - 1);
    const int y = p >> WS;
    const int xx = p & (W - 1);
    const uint32_t nbase = ((uint32_t)(bb * Cin)) << HWS;
    const uint32_t yx = ((uint32_t)y << WS) + (uint32_t)xx;

    v8f acc[2][2] = {{{}, {}}, {{}, {}}};

    v8h va8;
    v8h vblo = {}, vbhi = {};
    h16 okh = (h16)1.0f;        // chunk-constant border predicate (f16 0/1)

    auto load_tiles = [&](int k0) {
        // A tile (weights): K-contiguous, rows zero-padded -> one b128 load
        va8 = *(const v8h*)(wp + arow + (uint32_t)(k0 + ka));
        // B tile (im2col), single uniform (ky,kx) group per chunk
        uint32_t base;
        if constexpr (!K3) {
            base = nbase + (((uint32_t)(k0 + kb)) << HWS) + yx;
            okh = (h16)1.0f;
        } else {
            int g = k0 >> cin_shift;        // uniform: ky*3+kx
            int cib = k0 & (Cin - 1);       // uniform ci base of chunk
            int ky = g / 3, kx = g - ky * 3;
            int iy = y + ky - 1, ix = xx + kx - 1;
            bool ok = ((unsigned)iy < (unsigned)H) &
                      ((unsigned)ix < (unsigned)W);
            base = nbase + (((uint32_t)(cib + kb)) << HWS) +
                   ((uint32_t)iy << WS) + (uint32_t)ix;
            base = ok ? base : 0u;          // clamp to valid memory
            okh = ok ? (h16)1.0f : (h16)0.0f;
        }
#pragma unroll
        for (int r = 0; r < 16; ++r) {
            h16 v = in[base + ((uint32_t)r << HWS)];  // immediate offsets
            if (r < 8) vblo[r] = v; else vbhi[r - 8] = v;
        }
    };

    auto store_tiles = [&]() {
        *(v8h*)(Als + mi * 32 + ka) = va8;
        if constexpr (!K3) {
            *(v8h*)(Bls + nl * 32 + kb) = vblo;
            *(v8h*)(Bls + nl * 32 + kb + 8) = vbhi;
        } else {
            // splat multiply: OOB threads loaded valid memory * 0 -> 0
            *(v8h*)(Bls + nl * 32 + kb) = vblo * okh;
            *(v8h*)(Bls + nl * 32 + kb + 8) = vbhi * okh;
        }
    };

    load_tiles(0);
    for (int k0 = 0; k0 < K; k0 += 32) {
        __syncthreads();
        store_tiles();
        __syncthreads();
        if (k0 + 32 < K) load_tiles(k0 + 32);  // prefetch overlaps wmma below

        const h16* ar0 = Als + (wr * 32 + (lane & 15)) * 32;
        v16h a0 = lds_load_a_frag(ar0, lane);
        v16h a1 = lds_load_a_frag(ar0 + 16 * 32, lane);
        int kbase = (lane >> 4) << 4;
        const h16* bp = Bls + (wc * 32 + (lane & 15)) * 32 + kbase;
        v16h b0 = *(const v16h*)bp;
        v16h b1 = *(const v16h*)(bp + 16 * 32);
        acc[0][0] = WMMA_F16(a0, b0, acc[0][0]);
        acc[0][1] = WMMA_F16(a0, b1, acc[0][1]);
        acc[1][0] = WMMA_F16(a1, b0, acc[1][0]);
        acc[1][1] = WMMA_F16(a1, b1, acc[1][1]);
    }

    // --- epilogue: fused BN + ReLU (iff scale), NCHW scatter ---
#pragma unroll
    for (int fi = 0; fi < 2; ++fi) {
        int mbase = m0 + wr * 32 + fi * 16 + ((lane >> 4) << 3);
#pragma unroll
        for (int r = 0; r < 8; ++r) {
            int m = mbase + r;
            if constexpr (MGUARD) {
                if (m >= Cout) continue;
            }
#pragma unroll
            for (int sub = 0; sub < 2; ++sub) {
                int n = n0 + wc * 32 + sub * 16 + (lane & 15);
                float v = acc[fi][sub][r];
                if (scale) {  // BN + ReLU convs
                    v = v * scale[m] + shift[m];
                    v = v > 0.0f ? v : 0.0f;
                }
                int obb = n >> HWS;
                int op = n & (HW - 1);
                uint32_t idx = (uint32_t)(((obb * Cout + m) << HWS) + op);
                if (out16) out16[idx] = (h16)v;
                else       out32[idx] = v;
            }
        }
    }
}

// ---------------------------------------------------------------------------
// General-stride f16 GEMM (f32 accumulate): C[m,n] = sum_k A[m,k]*B[k,n]
// REQUIRES M%64==0, N%128==0, K%32==0 (all call sites satisfy this) ->
// completely guard-free. AK1/BK1: operand has K-stride 1 -> v8h loads.
// ---------------------------------------------------------------------------
template <bool AK1, bool BK1>
__global__ __launch_bounds__(256)
void gemm_wmma(const h16* __restrict__ A, int a_rs, int a_cs, int a_bs,
               const h16* __restrict__ B, int b_rs, int b_cs, int b_bs,
               float* __restrict__ C32, h16* __restrict__ C16,
               int c_ld, int c_bs, int K) {
    __shared__ h16 Als[64 * 32] __attribute__((aligned(32)));
    __shared__ h16 Bls[128 * 32] __attribute__((aligned(32)));

    const int t = threadIdx.x;
    const int lane = t & 31;
    const int wave = t >> 5;
    const int wr = wave >> 2, wc = wave & 3;
    const int n0 = blockIdx.x * 128;
    const int m0 = blockIdx.y * 64;
    const h16* Ab = A + (size_t)blockIdx.z * (size_t)a_bs;
    const h16* Bb = B + (size_t)blockIdx.z * (size_t)b_bs;

    const int mi = t & 63;
    const int ka = (t >> 6) << 3;
    const int nl = t & 127;
    const int kb = (t >> 7) << 4;
    const uint32_t abase = (uint32_t)((m0 + mi) * a_rs);
    const uint32_t bbase = (uint32_t)((n0 + nl) * b_cs);

    v8f acc[2][2] = {{{}, {}}, {{}, {}}};
    v8h va8 = {};
    v8h vblo = {}, vbhi = {};

    auto load_tiles = [&](int k0) {
        if constexpr (AK1) {
            va8 = *(const v8h*)(Ab + abase + (uint32_t)(k0 + ka));
        } else {
#pragma unroll
            for (int r = 0; r < 8; ++r)
                va8[r] = Ab[abase + (uint32_t)((k0 + ka + r) * a_cs)];
        }
        if constexpr (BK1) {
            vblo = *(const v8h*)(Bb + bbase + (uint32_t)(k0 + kb));
            vbhi = *(const v8h*)(Bb + bbase + (uint32_t)(k0 + kb + 8));
        } else {
#pragma unroll
            for (int r = 0; r < 16; ++r) {
                h16 v = Bb[(uint32_t)((k0 + kb + r) * b_rs) + bbase];
                if (r < 8) vblo[r] = v; else vbhi[r - 8] = v;
            }
        }
    };

    auto store_tiles = [&]() {
        *(v8h*)(Als + mi * 32 + ka) = va8;
        *(v8h*)(Bls + nl * 32 + kb) = vblo;
        *(v8h*)(Bls + nl * 32 + kb + 8) = vbhi;
    };

    load_tiles(0);
    for (int k0 = 0; k0 < K; k0 += 32) {
        __syncthreads();
        store_tiles();
        __syncthreads();
        if (k0 + 32 < K) load_tiles(k0 + 32);

        const h16* ar0 = Als + (wr * 32 + (lane & 15)) * 32;
        v16h a0 = lds_load_a_frag(ar0, lane);
        v16h a1 = lds_load_a_frag(ar0 + 16 * 32, lane);
        int kbase = (lane >> 4) << 4;
        const h16* bp = Bls + (wc * 32 + (lane & 15)) * 32 + kbase;
        v16h b0 = *(const v16h*)bp;
        v16h b1 = *(const v16h*)(bp + 16 * 32);
        acc[0][0] = WMMA_F16(a0, b0, acc[0][0]);
        acc[0][1] = WMMA_F16(a0, b1, acc[0][1]);
        acc[1][0] = WMMA_F16(a1, b0, acc[1][0]);
        acc[1][1] = WMMA_F16(a1, b1, acc[1][1]);
    }

    float* Cb32 = C32 ? C32 + (size_t)blockIdx.z * (size_t)c_bs : nullptr;
    h16*   Cb16 = C16 ? C16 + (size_t)blockIdx.z * (size_t)c_bs : nullptr;
#pragma unroll
    for (int fi = 0; fi < 2; ++fi) {
        int mbase = m0 + wr * 32 + fi * 16 + ((lane >> 4) << 3);
#pragma unroll
        for (int r = 0; r < 8; ++r) {
            int m = mbase + r;
#pragma unroll
            for (int sub = 0; sub < 2; ++sub) {
                int n = n0 + wc * 32 + sub * 16 + (lane & 15);
                uint32_t idx = (uint32_t)(m * c_ld + n);
                float v = acc[fi][sub][r];
                if (Cb32) Cb32[idx] = v;
                else      Cb16[idx] = (h16)v;
            }
        }
    }
}

// ---------------------------------------------------------------------------
// Row softmax: one 256-thread block per row; f32 in, f16 out (values <= 1).
// ---------------------------------------------------------------------------
__global__ __launch_bounds__(256)
void softmax_rows(const float* __restrict__ L, h16* __restrict__ S, int nc) {
    __shared__ float red[256];
    const int t = threadIdx.x;
    const size_t row = blockIdx.x;
    const float* lp = L + row * (size_t)nc;
    h16* sp = S + row * (size_t)nc;

    float mx = -3.4e38f;
    for (int c = t; c < nc; c += 256) mx = fmaxf(mx, lp[c]);
    red[t] = mx;
    __syncthreads();
    for (int s = 128; s > 0; s >>= 1) {
        if (t < s) red[t] = fmaxf(red[t], red[t + s]);
        __syncthreads();
    }
    mx = red[0];
    __syncthreads();

    float sum = 0.0f;
    for (int c = t; c < nc; c += 256) {
        float e = __expf(lp[c] - mx);
        sp[c] = (h16)e;
        sum += e;
    }
    red[t] = sum;
    __syncthreads();
    for (int s = 128; s > 0; s >>= 1) {
        if (t < s) red[t] += red[t + s];
        __syncthreads();
    }
    float inv = 1.0f / red[0];
    __syncthreads();
    for (int c = t; c < nc; c += 256) sp[c] = (h16)((float)sp[c] * inv);
}

// --------------------------- pointwise helpers -----------------------------

__global__ void fold_bn_kernel(const float* g, const float* b, const float* m,
                               const float* v, float* scale, float* shift,
                               int C) {
    int i = blockIdx.x * blockDim.x + threadIdx.x;
    if (i < C) {
        float s = g[i] * rsqrtf(v[i] + BN_EPS);
        scale[i] = s;
        shift[i] = b[i] - m[i] * s;
    }
}

__global__ void cvt_f32_f16_kernel(const float* __restrict__ in,
                                   h16* __restrict__ out, size_t n) {
    size_t i = blockIdx.x * (size_t)blockDim.x + threadIdx.x;
    if (i < n) out[i] = (h16)in[i];
}

// Conv weight repack: OIHW f32 -> wp[co*K + (ky*3+kx)*Cin + ci] f16,
// rows zero-padded to Cout_pad. kk = 9 (3x3) or 1 (1x1, identity perm).
__global__ void pack_w_kernel(const float* __restrict__ w,
                              h16* __restrict__ wp, int Cout, int Cin,
                              int cin_mask, int kk, size_t total) {
    size_t i = blockIdx.x * (size_t)blockDim.x + threadIdx.x;
    if (i >= total) return;
    int K = Cin * kk;
    int co = (int)(i / (unsigned)K);
    int rem = (int)(i - (size_t)co * (unsigned)K);
    h16 v = (h16)0.0f;
    if (co < Cout) {
        if (kk == 9) {
            int g = rem / Cin;              // Cin power of two; cheap shift
            int ci = rem & cin_mask;
            v = (h16)w[((size_t)co * Cin + ci) * 9 + g];
        } else {
            v = (h16)w[i];
        }
    }
    wp[i] = v;
}

// pa + ca = (gamma*E + x_pa) + (beta*X + x_ca)
__global__ void combine_attn_kernel(const h16* __restrict__ E,
                                    const h16* __restrict__ xpa,
                                    const h16* __restrict__ X,
                                    const h16* __restrict__ xca,
                                    const float* __restrict__ gamma,
                                    const float* __restrict__ beta,
                                    h16* __restrict__ out, size_t n) {
    size_t i = blockIdx.x * (size_t)blockDim.x + threadIdx.x;
    if (i < n) {
        float g = gamma[0], bt = beta[0];
        float v = g * (float)E[i] + (float)xpa[i] + bt * (float)X[i] +
                  (float)xca[i];
        out[i] = (h16)v;
    }
}

// 8x bilinear upsample, align_corners (32x32 -> 256x256), f16 NCHW planes.
__global__ void upsample8_kernel(const h16* __restrict__ in,
                                 h16* __restrict__ out, int planes) {
    size_t i = blockIdx.x * (size_t)blockDim.x + threadIdx.x;
    size_t total = (size_t)planes << 16;
    if (i >= total) return;
    int pc = (int)(i >> 16);
    int p = (int)(i & 65535);
    int oy = p >> 8, ox = p & 255;
    float fy = oy * (31.0f / 255.0f);
    float fx = ox * (31.0f / 255.0f);
    int y0 = (int)fy, x0 = (int)fx;
    float wy = fy - y0, wx = fx - x0;
    int y1 = y0 + 1 > 31 ? 31 : y0 + 1;
    int x1 = x0 + 1 > 31 ? 31 : x0 + 1;
    const h16* ip = in + ((size_t)pc << 10);
    float v00 = (float)ip[(y0 << 5) + x0];
    float v01 = (float)ip[(y0 << 5) + x1];
    float v10 = (float)ip[(y1 << 5) + x0];
    float v11 = (float)ip[(y1 << 5) + x1];
    float r0 = v00 + (v01 - v00) * wx;
    float r1 = v10 + (v11 - v10) * wx;
    out[i] = (h16)(r0 + (r1 - r0) * wy);
}

// ---------------------------------------------------------------------------

static inline size_t align256(size_t x) { return (x + 255) & ~(size_t)255; }

extern "C" void kernel_launch(void* const* d_in, const int* in_sizes, int n_in,
                              void* d_out, int out_size, void* d_ws,
                              size_t ws_size, hipStream_t stream) {
    (void)in_sizes; (void)n_in; (void)out_size; (void)ws_size;

    const int B = 4, Cin = 2048, C4 = 512, C8 = 256, NC = 19;
    const int HW = 1024;       // 32*32
    const int HW2 = 65536;     // 256*256
    const int NCpad = 64;      // conv4b output channels padded to tile M

    const float* x_f32   = (const float*)d_in[0];
    const float* w1      = (const float*)d_in[1];
    const float* w2      = (const float*)d_in[6];
    const float* wB      = (const float*)d_in[11];
    const float* gamma_p = (const float*)d_in[12];
    const float* beta_p  = (const float*)d_in[13];
    const float* w3      = (const float*)d_in[14];
    const float* w4a     = (const float*)d_in[19];
    const float* w4b     = (const float*)d_in[24];
    float* out = (float*)d_out;

    // workspace carve-out
    char* ws = (char*)d_ws;
    size_t off = 0;
    auto alloc = [&](size_t bytes) -> void* {
        void* p = (void*)(ws + off);
        off = align256(off + bytes);
        return p;
    };

    const size_t nx = (size_t)B * Cin * HW;            // 8,388,608
    const size_t nact = (size_t)B * C4 * HW;           // 2,097,152
    h16* x16   = (h16*)alloc(nx * 2);
    h16* w1p   = (h16*)alloc((size_t)C4 * Cin * 9 * 2);
    h16* w2p   = (h16*)alloc((size_t)C4 * Cin * 9 * 2);
    h16* wBp   = (h16*)alloc((size_t)C4 * C4 * 2);
    h16* w3p   = (h16*)alloc((size_t)C4 * C4 * 9 * 2);
    h16* w4ap  = (h16*)alloc((size_t)C8 * C4 * 9 * 2);
    h16* w4bp  = (h16*)alloc((size_t)NCpad * C8 * 9 * 2);
    float* s1 = (float*)alloc(C4 * 4); float* h1 = (float*)alloc(C4 * 4);
    float* s2 = (float*)alloc(C4 * 4); float* h2 = (float*)alloc(C4 * 4);
    float* s3 = (float*)alloc(C4 * 4); float* h3 = (float*)alloc(C4 * 4);
    float* s4 = (float*)alloc(C8 * 4); float* h4 = (float*)alloc(C8 * 4);
    h16* xpa16 = (h16*)alloc(nact * 2);
    h16* xca16 = (h16*)alloc(nact * 2);
    h16* bm16  = (h16*)alloc(nact * 2);
    float* Lbuf = (float*)alloc((size_t)B * HW * HW * 4);   // 16.8 MB
    h16* S16   = (h16*)alloc((size_t)B * HW * HW * 2);      // 8.4 MB
    h16* E16   = (h16*)alloc(nact * 2);
    float* CL  = (float*)alloc((size_t)B * C4 * C4 * 4);
    h16* CS16  = (h16*)alloc((size_t)B * C4 * C4 * 2);
    h16* X16   = (h16*)alloc(nact * 2);
    h16* sum16 = (h16*)alloc(nact * 2);
    h16* y316  = (h16*)alloc(nact * 2);
    h16* up16  = (h16*)alloc((size_t)B * C4 * HW2 * 2);     // 268 MB
    h16* y4a16 = (h16*)alloc((size_t)B * C8 * HW2 * 2);     // 134 MB

    // --- BN folds ---
    fold_bn_kernel<<<2, 256, 0, stream>>>((const float*)d_in[2],
        (const float*)d_in[3], (const float*)d_in[4], (const float*)d_in[5],
        s1, h1, C4);
    fold_bn_kernel<<<2, 256, 0, stream>>>((const float*)d_in[7],
        (const float*)d_in[8], (const float*)d_in[9], (const float*)d_in[10],
        s2, h2, C4);
    fold_bn_kernel<<<2, 256, 0, stream>>>((const float*)d_in[15],
        (const float*)d_in[16], (const float*)d_in[17], (const float*)d_in[18],
        s3, h3, C4);
    fold_bn_kernel<<<1, 256, 0, stream>>>((const float*)d_in[20],
        (const float*)d_in[21], (const float*)d_in[22], (const float*)d_in[23],
        s4, h4, C8);

    // --- f16 conversions / weight repacking ((ky,kx,ci) K-order) ---
    cvt_f32_f16_kernel<<<(unsigned)((nx + 255) / 256), 256, 0, stream>>>(
        x_f32, x16, nx);
    {
        size_t n1 = (size_t)C4 * Cin * 9;
        pack_w_kernel<<<(unsigned)((n1 + 255) / 256), 256, 0, stream>>>(
            w1, w1p, C4, Cin, Cin - 1, 9, n1);
        pack_w_kernel<<<(unsigned)((n1 + 255) / 256), 256, 0, stream>>>(
            w2, w2p, C4, Cin, Cin - 1, 9, n1);
        size_t nB = (size_t)C4 * C4;
        pack_w_kernel<<<(unsigned)((nB + 255) / 256), 256, 0, stream>>>(
            wB, wBp, C4, C4, C4 - 1, 1, nB);
        size_t n3 = (size_t)C4 * C4 * 9;
        pack_w_kernel<<<(unsigned)((n3 + 255) / 256), 256, 0, stream>>>(
            w3, w3p, C4, C4, C4 - 1, 9, n3);
        size_t n4a = (size_t)C8 * C4 * 9;
        pack_w_kernel<<<(unsigned)((n4a + 255) / 256), 256, 0, stream>>>(
            w4a, w4ap, C8, C4, C4 - 1, 9, n4a);
        size_t n4bp = (size_t)NCpad * C8 * 9;
        pack_w_kernel<<<(unsigned)((n4bp + 255) / 256), 256, 0, stream>>>(
            w4b, w4bp, NC, C8, C8 - 1, 9, n4bp);
    }

    // --- conv1 / conv2 (2048 -> 512, 3x3, BN+ReLU) + pa 1x1 conv ---
    {
        dim3 g((B * HW) / 128, C4 / 64);
        conv_implicit_wmma<false, true, 10, 5><<<g, 256, 0, stream>>>(
            x16, w1p, s1, h1, xpa16, nullptr, Cin, 11, C4);
        conv_implicit_wmma<false, true, 10, 5><<<g, 256, 0, stream>>>(
            x16, w2p, s2, h2, xca16, nullptr, Cin, 11, C4);
        conv_implicit_wmma<false, false, 10, 5><<<g, 256, 0, stream>>>(
            xpa16, wBp, nullptr, nullptr, bm16, nullptr, C4, 9, C4);
    }

    // --- position attention ---
    // L[n,m] = sum_c Bm[c,n]*Bm[c,m]  (per batch, 1024x1024, K=512)
    gemm_wmma<false, false><<<dim3(8, 16, B), 256, 0, stream>>>(
        bm16, 1, 1024, C4 * HW,
        bm16, 1024, 1, C4 * HW,
        Lbuf, nullptr, 1024, HW * HW, C4);
    softmax_rows<<<B * HW, 256, 0, stream>>>(Lbuf, S16, HW);
    // E[c,n] = sum_m Bm[c,m]*S[n,m]  (A and B both K-contiguous)
    gemm_wmma<true, true><<<dim3(8, 8, B), 256, 0, stream>>>(
        bm16, 1024, 1, C4 * HW,
        S16, 1, 1024, HW * HW,
        nullptr, E16, 1024, C4 * HW, HW);

    // --- channel attention ---
    // CL[c,d] = sum_n xf[c,n]*xf[d,n]  (512x512, K=1024; both K-contiguous)
    gemm_wmma<true, true><<<dim3(4, 8, B), 256, 0, stream>>>(
        xca16, 1024, 1, C4 * HW,
        xca16, 1, 1024, C4 * HW,
        CL, nullptr, 512, C4 * C4, HW);
    softmax_rows<<<B * C4, 256, 0, stream>>>(CL, CS16, C4);
    // X[c,n] = sum_k CS[k,c]*xf[k,n]
    gemm_wmma<false, false><<<dim3(8, 8, B), 256, 0, stream>>>(
        CS16, 1, 512, C4 * C4,
        xca16, 1024, 1, C4 * HW,
        nullptr, X16, 1024, C4 * HW, C4);

    // --- combine: gamma*E + x_pa + beta*X + x_ca ---
    combine_attn_kernel<<<(unsigned)((nact + 255) / 256), 256, 0, stream>>>(
        E16, xpa16, X16, xca16, gamma_p, beta_p, sum16, nact);

    // --- conv3 (512 -> 512, 3x3, BN+ReLU) ---
    {
        dim3 g((B * HW) / 128, C4 / 64);
        conv_implicit_wmma<false, true, 10, 5><<<g, 256, 0, stream>>>(
            sum16, w3p, s3, h3, y316, nullptr, C4, 9, C4);
    }

    // --- 8x bilinear upsample (align_corners) ---
    {
        size_t nup = (size_t)B * C4 * HW2;
        upsample8_kernel<<<(unsigned)((nup + 255) / 256), 256, 0, stream>>>(
            y316, up16, B * C4);
    }

    // --- conv4a (512 -> 256, 3x3, BN+ReLU) on 256x256 ---
    {
        dim3 g((B * HW2) / 128, C8 / 64);
        conv_implicit_wmma<false, true, 16, 8><<<g, 256, 0, stream>>>(
            up16, w4ap, s4, h4, y4a16, nullptr, C4, 9, C8);
    }

    // --- conv4b (256 -> 19, 3x3), f32 output, channel-guarded stores ---
    {
        dim3 g((B * HW2) / 128, 1);
        conv_implicit_wmma<true, true, 16, 8><<<g, 256, 0, stream>>>(
            y4a16, w4bp, nullptr, nullptr, nullptr, out, C8, 8, NC);
    }
}
